// GatedDeltaRuleRecurrent_34720515621543
// MI455X (gfx1250) — compile-verified, matrix operand order checked by reading, not run on hardware
//
#include <hip/hip_runtime.h>

#define D 512
#define BS 16384   // B*S = 8*2048
#define SEQ 2048

typedef __attribute__((ext_vector_type(16))) __bf16 v16bf;
typedef __attribute__((ext_vector_type(8)))  __bf16 v8bf;
typedef __attribute__((ext_vector_type(8)))  float  v8f;

static __device__ __forceinline__ __bf16 to_bf16(float f) {
    unsigned u = __builtin_bit_cast(unsigned, f);
    u += 0x7FFFu + ((u >> 16) & 1u);           // round-to-nearest-even
    unsigned short h = (unsigned short)(u >> 16);
    return __builtin_bit_cast(__bf16, h);
}

static __device__ __forceinline__ float sigmoidf_(float x) {
    return 1.0f / (1.0f + __expf(-x));
}

// ---------------------------------------------------------------------------
// f32 -> bf16 bulk conversion (one b128 store per thread).
// ---------------------------------------------------------------------------
__global__ void __launch_bounds__(256)
cvt_f32_bf16(const float* __restrict__ in, __bf16* __restrict__ out, int n) {
    const int i = (blockIdx.x * 256 + threadIdx.x) * 8;
    if (i >= n) return;
    v8bf o;
#pragma unroll
    for (int e = 0; e < 8; ++e) o[e] = to_bf16(in[i + e]);
    *(v8bf*)(out + i) = o;
}

// ---------------------------------------------------------------------------
// Y = Xb @ Wb^T + bias.  Xb: (M, D) bf16, Wb: (D, D) bf16 (N-major), Y f32.
// One wave computes 64x32 output: 4 M-tiles x 2 N-tiles = 8 WMMA accumulators.
// Per K-step: 12 x b128 loads feed 8 x v_wmma_f32_16x16x32_bf16.
// Block = 128 threads (4 waves, split along N); grid = (M/64, D/128).
// ---------------------------------------------------------------------------
__global__ void __launch_bounds__(128)
gemm_bf16_wmma(const __bf16* __restrict__ Xb, const __bf16* __restrict__ Wb,
               const float* __restrict__ bias, float* __restrict__ Y) {
    const int lane = threadIdx.x & 31;
    const int wave = threadIdx.x >> 5;
    const int l15  = lane & 15;
    const bool hi  = lane >= 16;

    const int m0 = blockIdx.x * 64;
    const int n0 = blockIdx.y * 128 + wave * 32;

    const __bf16* xr0 = Xb + (size_t)(m0 + 0 * 16 + l15) * D;
    const __bf16* xr1 = Xb + (size_t)(m0 + 1 * 16 + l15) * D;
    const __bf16* xr2 = Xb + (size_t)(m0 + 2 * 16 + l15) * D;
    const __bf16* xr3 = Xb + (size_t)(m0 + 3 * 16 + l15) * D;
    const __bf16* wr0 = Wb + (size_t)(n0 + 0 * 16 + l15) * D;
    const __bf16* wr1 = Wb + (size_t)(n0 + 1 * 16 + l15) * D;

    v8f acc[4][2] = {};
    for (int k0 = 0; k0 < D; k0 += 32) {
        // 16-bit fragment layout: lanes<16 take K=[k0..k0+7]+[k0+16..23],
        // lanes>=16 take K=[k0+8..15]+[k0+24..31]; each half is one b128.
        const int ka = k0 + (hi ? 8 : 0);
        const int kb = k0 + (hi ? 24 : 16);

        v16bf af[4], bfr[2];
        {
            v8bf lo, hh;
            lo = *(const v8bf*)(xr0 + ka); hh = *(const v8bf*)(xr0 + kb);
            af[0] = __builtin_shufflevector(lo, hh, 0,1,2,3,4,5,6,7,8,9,10,11,12,13,14,15);
            lo = *(const v8bf*)(xr1 + ka); hh = *(const v8bf*)(xr1 + kb);
            af[1] = __builtin_shufflevector(lo, hh, 0,1,2,3,4,5,6,7,8,9,10,11,12,13,14,15);
            lo = *(const v8bf*)(xr2 + ka); hh = *(const v8bf*)(xr2 + kb);
            af[2] = __builtin_shufflevector(lo, hh, 0,1,2,3,4,5,6,7,8,9,10,11,12,13,14,15);
            lo = *(const v8bf*)(xr3 + ka); hh = *(const v8bf*)(xr3 + kb);
            af[3] = __builtin_shufflevector(lo, hh, 0,1,2,3,4,5,6,7,8,9,10,11,12,13,14,15);
            lo = *(const v8bf*)(wr0 + ka); hh = *(const v8bf*)(wr0 + kb);
            bfr[0] = __builtin_shufflevector(lo, hh, 0,1,2,3,4,5,6,7,8,9,10,11,12,13,14,15);
            lo = *(const v8bf*)(wr1 + ka); hh = *(const v8bf*)(wr1 + kb);
            bfr[1] = __builtin_shufflevector(lo, hh, 0,1,2,3,4,5,6,7,8,9,10,11,12,13,14,15);
        }
#pragma unroll
        for (int i = 0; i < 4; ++i)
#pragma unroll
            for (int j = 0; j < 2; ++j)
                acc[i][j] = __builtin_amdgcn_wmma_f32_16x16x32_bf16(
                    false, af[i], false, bfr[j], (short)0, acc[i][j], false, false);
    }

#pragma unroll
    for (int j = 0; j < 2; ++j) {
        const int   col = n0 + j * 16 + l15;
        const float bn  = bias[col];
#pragma unroll
        for (int i = 0; i < 4; ++i) {
            const int mbase = m0 + i * 16 + (hi ? 8 : 0);
#pragma unroll
            for (int r = 0; r < 8; ++r)
                Y[(size_t)(mbase + r) * D + col] = acc[i][j][r] + bn;
        }
    }
}

// ---------------------------------------------------------------------------
// alpha = sigmoid(x @ Wa^T + ba), beta = sigmoid(x @ Wb^T + bb)
// ---------------------------------------------------------------------------
__global__ void __launch_bounds__(256)
gemv_ab(const float* __restrict__ X,
        const float* __restrict__ Wa, const float* __restrict__ ba,
        const float* __restrict__ Wb, const float* __restrict__ bb,
        float* __restrict__ alpha, float* __restrict__ beta) {
    const int row  = blockIdx.x * (blockDim.x >> 5) + (threadIdx.x >> 5);
    const int lane = threadIdx.x & 31;
    const float* xr = X + (size_t)row * D;
    float sa = 0.0f, sb = 0.0f;
    for (int j = lane; j < D; j += 32) {
        const float xv = xr[j];
        sa += xv * Wa[j];
        sb += xv * Wb[j];
    }
#pragma unroll
    for (int off = 16; off; off >>= 1) {
        sa += __shfl_down(sa, off);
        sb += __shfl_down(sb, off);
    }
    if (lane == 0) {
        alpha[row] = sigmoidf_(sa + ba[0]);
        beta[row]  = sigmoidf_(sb + bb[0]);
    }
}

// ---------------------------------------------------------------------------
// Causal depthwise conv (k=4) + bias + SiLU, optional row L2-norm.
// ---------------------------------------------------------------------------
__global__ void __launch_bounds__(256)
conv_silu(const float* __restrict__ Yin, const float* __restrict__ cw,
          const float* __restrict__ cb, float* __restrict__ Out, int do_norm) {
    const int bs  = blockIdx.x;
    const int s   = bs % SEQ;
    const int tid = threadIdx.x;
    __shared__ float red[256];

    float vals[2];
    float sq = 0.0f;
#pragma unroll
    for (int e = 0; e < 2; ++e) {
        const int c = tid * 2 + e;
        float acc = cb[c];
#pragma unroll
        for (int t = 0; t < 4; ++t) {
            const int sp = s - 3 + t;
            if (sp >= 0)
                acc += cw[c * 4 + t] * Yin[(size_t)(bs - 3 + t) * D + c];
        }
        const float sl = acc * sigmoidf_(acc);
        vals[e] = sl;
        sq += sl * sl;
    }

    float inv = 1.0f;
    if (do_norm) {
        red[tid] = sq;
        __syncthreads();
        for (int o = 128; o; o >>= 1) {
            if (tid < o) red[tid] += red[tid + o];
            __syncthreads();
        }
        inv = 1.0f / fmaxf(sqrtf(red[0]), 1e-12f);
    }
    Out[(size_t)bs * D + tid * 2]     = vals[0] * inv;
    Out[(size_t)bs * D + tid * 2 + 1] = vals[1] * inv;
}

// ---------------------------------------------------------------------------
// Gated delta rule recurrence; state rows independent, state in VGPRs.
// k_t/q_t staged to LDS with CDNA5 async global->LDS copies (b128/lane).
// Block = 256 thr = 64 rows (4 lanes x 128 cols per row); grid = (8, 8).
// ---------------------------------------------------------------------------
__global__ void __launch_bounds__(256)
gdn_recurrence(const float* __restrict__ q, const float* __restrict__ k,
               const float* __restrict__ v, const float* __restrict__ alpha,
               const float* __restrict__ beta, float* __restrict__ out) {
    extern __shared__ float smem[];       // kbuf[512] | qbuf[512]
    float* kbuf = smem;
    float* qbuf = smem + D;

    const int b   = blockIdx.x;
    const int rb  = blockIdx.y;
    const int tid = threadIdx.x;
    const int row = rb * 64 + (tid >> 2);
    const int c0  = (tid & 3) * 128;

    float st[128];
#pragma unroll
    for (int j = 0; j < 128; ++j) st[j] = 0.0f;

    const size_t base = (size_t)b * SEQ * D;
    for (int t = 0; t < SEQ; ++t) {
        const size_t roff = base + (size_t)t * D;
        __syncthreads();                  // prior-step consumers done
        {
            // 256 threads x 16B: bytes [0,2048) <- k_t, [2048,4096) <- q_t
            const float* src = (tid < 128) ? (k + roff + tid * 4)
                                           : (q + roff + (tid - 128) * 4);
            unsigned lds_off = (unsigned)(unsigned long long)(smem + tid * 4);
            asm volatile("global_load_async_to_lds_b128 %0, %1, off"
                         :: "v"(lds_off),
                            "v"((unsigned long long)(size_t)src)
                         : "memory");
            asm volatile("s_wait_asynccnt 0x0" ::: "memory");
        }
        __syncthreads();

        float skp = 0.0f;
#pragma unroll
        for (int j = 0; j < 128; ++j) skp += st[j] * kbuf[c0 + j];
        skp += __shfl_xor(skp, 1);
        skp += __shfl_xor(skp, 2);

        const float at  = alpha[b * SEQ + t];
        const float bt  = beta[b * SEQ + t];
        const float vr  = v[roff + row];
        const float bsk = bt * skp;

        float op = 0.0f;
#pragma unroll
        for (int j = 0; j < 128; ++j) {
            const float kj = kbuf[c0 + j];
            const float sn = at * (st[j] - bsk * kj) + bt * vr * kj;
            st[j] = sn;
            op += sn * qbuf[c0 + j];
        }
        op += __shfl_xor(op, 1);
        op += __shfl_xor(op, 2);
        if ((tid & 3) == 0) out[roff + row] = op;
    }
}

// ---------------------------------------------------------------------------
// z = zc_rmsnorm(gdn) * norm_w * silu(yg).  One block per row.
// ---------------------------------------------------------------------------
__global__ void __launch_bounds__(256)
rmsnorm_gate(const float* __restrict__ gdn, const float* __restrict__ yg,
             const float* __restrict__ nw, float* __restrict__ z) {
    const int bs  = blockIdx.x;
    const int tid = threadIdx.x;
    __shared__ float red[256];
    const size_t off = (size_t)bs * D;
    const int    c0  = tid * 2;

    const float x0 = gdn[off + c0], x1 = gdn[off + c0 + 1];
    red[tid] = x0 + x1;
    __syncthreads();
    for (int o = 128; o; o >>= 1) {
        if (tid < o) red[tid] += red[tid + o];
        __syncthreads();
    }
    const float mean = red[0] * (1.0f / (float)D);
    __syncthreads();

    const float d0 = x0 - mean, d1 = x1 - mean;
    red[tid] = d0 * d0 + d1 * d1;
    __syncthreads();
    for (int o = 128; o; o >>= 1) {
        if (tid < o) red[tid] += red[tid + o];
        __syncthreads();
    }
    const float rinv = rsqrtf(red[0] * (1.0f / (float)D) + 1e-5f);

    float g0 = yg[off + c0], g1 = yg[off + c0 + 1];
    g0 = g0 * sigmoidf_(g0);
    g1 = g1 * sigmoidf_(g1);
    z[off + c0]     = d0 * rinv * nw[c0] * g0;
    z[off + c0 + 1] = d1 * rinv * nw[c0 + 1] * g1;
}

// ---------------------------------------------------------------------------
extern "C" void kernel_launch(void* const* d_in, const int* in_sizes, int n_in,
                              void* d_out, int out_size, void* d_ws, size_t ws_size,
                              hipStream_t stream) {
    const float* x      = (const float*)d_in[0];
    const float* Wq     = (const float*)d_in[1];
    const float* bq     = (const float*)d_in[2];
    const float* Wk     = (const float*)d_in[3];
    const float* bk     = (const float*)d_in[4];
    const float* Wv     = (const float*)d_in[5];
    const float* bv     = (const float*)d_in[6];
    const float* Wa     = (const float*)d_in[7];
    const float* ba     = (const float*)d_in[8];
    const float* Wb     = (const float*)d_in[9];
    const float* bb     = (const float*)d_in[10];
    const float* cwq    = (const float*)d_in[11];
    const float* cbq    = (const float*)d_in[12];
    const float* cwk    = (const float*)d_in[13];
    const float* cbk    = (const float*)d_in[14];
    const float* cwv    = (const float*)d_in[15];
    const float* cbv    = (const float*)d_in[16];
    const float* norm_w = (const float*)d_in[17];
    const float* Wg     = (const float*)d_in[18];
    const float* bg     = (const float*)d_in[19];
    const float* Wo     = (const float*)d_in[20];
    const float* bo     = (const float*)d_in[21];

    const size_t NBD = (size_t)BS * D;
    const size_t WSZ = (size_t)D * D;
    float* ws    = (float*)d_ws;
    float* yq    = ws + 0 * NBD;
    float* yk    = ws + 1 * NBD;
    float* yv    = ws + 2 * NBD;
    float* yg    = ws + 3 * NBD;
    float* qb    = ws + 4 * NBD;
    float* kb    = ws + 5 * NBD;
    float* vb    = ws + 6 * NBD;
    float* alpha = ws + 7 * NBD;
    float* beta  = alpha + BS;
    __bf16* xb   = (__bf16*)(beta + BS);
    __bf16* wqb  = xb + NBD;
    __bf16* wkb  = wqb + WSZ;
    __bf16* wvb  = wkb + WSZ;
    __bf16* wgb  = wvb + WSZ;
    __bf16* wob  = wgb + WSZ;
    float* gdn   = yq;                     // yq dead after conv
    float* z     = yk;                     // yk dead after conv
    __bf16* zb   = (__bf16*)yv;            // yv dead after conv

    // Precision staging: one-time f32 -> bf16 conversions.
    cvt_f32_bf16<<<NBD / 2048, 256, 0, stream>>>(x, xb, (int)NBD);
    cvt_f32_bf16<<<WSZ / 2048, 256, 0, stream>>>(Wq, wqb, (int)WSZ);
    cvt_f32_bf16<<<WSZ / 2048, 256, 0, stream>>>(Wk, wkb, (int)WSZ);
    cvt_f32_bf16<<<WSZ / 2048, 256, 0, stream>>>(Wv, wvb, (int)WSZ);
    cvt_f32_bf16<<<WSZ / 2048, 256, 0, stream>>>(Wg, wgb, (int)WSZ);
    cvt_f32_bf16<<<WSZ / 2048, 256, 0, stream>>>(Wo, wob, (int)WSZ);

    const dim3 gemmGrid(BS / 64, D / 128);  // (256, 4), 128 threads = 4 waves

    gemm_bf16_wmma<<<gemmGrid, 128, 0, stream>>>(xb, wqb, bq, yq);
    gemm_bf16_wmma<<<gemmGrid, 128, 0, stream>>>(xb, wkb, bk, yk);
    gemm_bf16_wmma<<<gemmGrid, 128, 0, stream>>>(xb, wvb, bv, yv);
    gemm_bf16_wmma<<<gemmGrid, 128, 0, stream>>>(xb, wgb, bg, yg);

    gemv_ab<<<BS / 8, 256, 0, stream>>>(x, Wa, ba, Wb, bb, alpha, beta);

    conv_silu<<<BS, 256, 0, stream>>>(yq, cwq, cbq, qb, 1);
    conv_silu<<<BS, 256, 0, stream>>>(yk, cwk, cbk, kb, 1);
    conv_silu<<<BS, 256, 0, stream>>>(yv, cwv, cbv, vb, 0);

    gdn_recurrence<<<dim3(8, 8), 256, 2 * D * sizeof(float), stream>>>(
        qb, kb, vb, alpha, beta, gdn);

    rmsnorm_gate<<<BS, 256, 0, stream>>>(gdn, yg, norm_w, z);

    cvt_f32_bf16<<<NBD / 2048, 256, 0, stream>>>(z, zb, (int)NBD);
    gemm_bf16_wmma<<<gemmGrid, 128, 0, stream>>>(zb, wob, bo, (float*)d_out);
}
